// SelfAttentionBlock_27298812133994
// MI455X (gfx1250) — matrix operand order repaired and checked by reading, not compiled, not run on hardware
//
#include <hip/hip_runtime.h>
#include <math.h>

#define D_MODEL 1024
#define N_HEADS 16
#define DH      64
#define BB      4
#define TT      2048

typedef __attribute__((ext_vector_type(16))) __bf16 v16bf;
typedef __attribute__((ext_vector_type(8)))  __bf16 v8bf;
typedef __attribute__((ext_vector_type(8)))  float  v8f;

static __device__ __forceinline__ v8f wmma_bf16(v16bf a, v16bf b, v8f c) {
  // D = A(16x32 bf16) * B(32x16 bf16) + C(16x16 f32)
  return __builtin_amdgcn_wmma_f32_16x16x32_bf16(false, a, false, b, (short)0, c,
                                                 false, false);
}

static __device__ __forceinline__ v16bf cat8(v8bf lo, v8bf hi) {
  v16bf a;
#pragma unroll
  for (int i = 0; i < 8; ++i) { a[i] = lo[i]; a[8 + i] = hi[i]; }
  return a;
}

// ---- A fragment (16x32, MxK), bf16 row-major source (ISA 7.12.2):
// lanes 0-15 (row m): K{0..7} in elems 0-7, K{16..23} in elems 8-15
// lanes16-31 (row m): K{8..15}, K{24..31}
static __device__ __forceinline__ v16bf load_A_bf16(const __bf16* base, int ld, int lane) {
  const int m = lane & 15, kh = lane >> 4;
  const __bf16* p = base + (size_t)m * ld + kh * 8;
  return cat8(*(const v8bf*)p, *(const v8bf*)(p + 16));
}

// ---- B fragment (32x16, KxN): lane holds column n=lane&15,
// K = (lane>>4)*8 + 0..7 in elems 0-7, 16 + (lane>>4)*8 + 0..7 in elems 8-15 (matches A).
static __device__ __forceinline__ v16bf load_B_bf16(const __bf16* base, int ld, int lane) {
  const int n = lane & 15, ks = (lane >> 4) * 8;
  const __bf16* p = base + (size_t)n * ld + ks;
  return cat8(*(const v8bf*)p, *(const v8bf*)(p + 16));
}

// =====================================================================
// Kernel 0: one-shot f32 -> bf16 conversion (x and the three W matrices).
// =====================================================================
__global__ __launch_bounds__(256)
void cvt_bf16_kernel(const float* __restrict__ src, __bf16* __restrict__ dst, int n) {
  const int i = (blockIdx.x * 256 + threadIdx.x) * 8;
  if (i + 8 <= n) {
    float4 f0 = *(const float4*)(src + i);
    float4 f1 = *(const float4*)(src + i + 4);
    v8bf o;
    o[0]=(__bf16)f0.x; o[1]=(__bf16)f0.y; o[2]=(__bf16)f0.z; o[3]=(__bf16)f0.w;
    o[4]=(__bf16)f1.x; o[5]=(__bf16)f1.y; o[6]=(__bf16)f1.z; o[7]=(__bf16)f1.w;
    *(v8bf*)(dst + i) = o;
  }
}

// =====================================================================
// Kernel 1: QKV projection, all-bf16, ping-pong double-buffered K-loop
// (two named fragment sets swap roles -> no register rotation copies).
// One wave: 32 tokens x 64 features (one head). q pre-scaled by 1/8.
// q,k -> [b,h,t,dh]; v -> transposed [b,h,dh,T].
// =====================================================================
__global__ __launch_bounds__(32)
void qkv_proj_kernel(const __bf16* __restrict__ xb,
                     const __bf16* __restrict__ Wqb, const float* __restrict__ bq,
                     const __bf16* __restrict__ Wkb, const float* __restrict__ bk,
                     const __bf16* __restrict__ Wvb, const float* __restrict__ bv,
                     __bf16* __restrict__ qb, __bf16* __restrict__ kb,
                     __bf16* __restrict__ vT) {
  const int lane  = threadIdx.x;
  const int which = blockIdx.z;             // 0:q 1:k 2:v
  const int head  = blockIdx.y;
  const int flat0 = blockIdx.x * 32;        // flattened b*T + t  (32 rows/wave)
  const int b  = flat0 / TT;
  const int t0 = flat0 % TT;
  const int n0 = head * DH;

  const __bf16* W    = (which == 0) ? Wqb : (which == 1) ? Wkb : Wvb;
  const float*  bias = (which == 0) ? bq  : (which == 1) ? bk  : bv;
  const __bf16* W0 = W + (size_t)(n0 +  0) * D_MODEL;
  const __bf16* W1 = W + (size_t)(n0 + 16) * D_MODEL;
  const __bf16* W2 = W + (size_t)(n0 + 32) * D_MODEL;
  const __bf16* W3 = W + (size_t)(n0 + 48) * D_MODEL;
  const __bf16* xrow0 = xb + (size_t)flat0 * D_MODEL;
  const __bf16* xrow1 = xb + (size_t)(flat0 + 16) * D_MODEL;

  v8f acc[2][4];
#pragma unroll
  for (int mt = 0; mt < 2; ++mt)
#pragma unroll
    for (int j = 0; j < 4; ++j)
#pragma unroll
      for (int i = 0; i < 8; ++i) acc[mt][j][i] = 0.f;

  auto wmma8 = [&](const v16bf& a0, const v16bf& a1, const v16bf& b0,
                   const v16bf& b1, const v16bf& b2, const v16bf& b3) {
    acc[0][0] = wmma_bf16(a0, b0, acc[0][0]);
    acc[1][0] = wmma_bf16(a1, b0, acc[1][0]);
    acc[0][1] = wmma_bf16(a0, b1, acc[0][1]);
    acc[1][1] = wmma_bf16(a1, b1, acc[1][1]);
    acc[0][2] = wmma_bf16(a0, b2, acc[0][2]);
    acc[1][2] = wmma_bf16(a1, b2, acc[1][2]);
    acc[0][3] = wmma_bf16(a0, b3, acc[0][3]);
    acc[1][3] = wmma_bf16(a1, b3, acc[1][3]);
  };

  // set A preload @ k = 0
  v16bf Aa0 = load_A_bf16(xrow0, D_MODEL, lane);
  v16bf Aa1 = load_A_bf16(xrow1, D_MODEL, lane);
  v16bf Ab0 = load_B_bf16(W0, D_MODEL, lane);
  v16bf Ab1 = load_B_bf16(W1, D_MODEL, lane);
  v16bf Ab2 = load_B_bf16(W2, D_MODEL, lane);
  v16bf Ab3 = load_B_bf16(W3, D_MODEL, lane);
  v16bf Ba0, Ba1, Bb0, Bb1, Bb2, Bb3;

#pragma unroll 1
  for (int it = 0; it < (D_MODEL / 64) - 1; ++it) {     // 15 double-steps
    const int kB = it * 64 + 32;
    Ba0 = load_A_bf16(xrow0 + kB, D_MODEL, lane);
    Ba1 = load_A_bf16(xrow1 + kB, D_MODEL, lane);
    Bb0 = load_B_bf16(W0 + kB, D_MODEL, lane);
    Bb1 = load_B_bf16(W1 + kB, D_MODEL, lane);
    Bb2 = load_B_bf16(W2 + kB, D_MODEL, lane);
    Bb3 = load_B_bf16(W3 + kB, D_MODEL, lane);
    __builtin_prefetch(xrow0 + kB + 64, 0, 1);          // global_prefetch_b8
    wmma8(Aa0, Aa1, Ab0, Ab1, Ab2, Ab3);                // consumes set A (k)
    const int kA = kB + 32;
    Aa0 = load_A_bf16(xrow0 + kA, D_MODEL, lane);
    Aa1 = load_A_bf16(xrow1 + kA, D_MODEL, lane);
    Ab0 = load_B_bf16(W0 + kA, D_MODEL, lane);
    Ab1 = load_B_bf16(W1 + kA, D_MODEL, lane);
    Ab2 = load_B_bf16(W2 + kA, D_MODEL, lane);
    Ab3 = load_B_bf16(W3 + kA, D_MODEL, lane);
    wmma8(Ba0, Ba1, Bb0, Bb1, Bb2, Bb3);                // consumes set B (k+32)
  }
  {                                                     // epilogue: k = 960, 992
    const int kB = D_MODEL - 32;
    Ba0 = load_A_bf16(xrow0 + kB, D_MODEL, lane);
    Ba1 = load_A_bf16(xrow1 + kB, D_MODEL, lane);
    Bb0 = load_B_bf16(W0 + kB, D_MODEL, lane);
    Bb1 = load_B_bf16(W1 + kB, D_MODEL, lane);
    Bb2 = load_B_bf16(W2 + kB, D_MODEL, lane);
    Bb3 = load_B_bf16(W3 + kB, D_MODEL, lane);
    wmma8(Aa0, Aa1, Ab0, Ab1, Ab2, Ab3);
    wmma8(Ba0, Ba1, Bb0, Bb1, Bb2, Bb3);
  }

  const int kh = lane >> 4;
  const int nl = lane & 15;
  const float oscale = (which == 0) ? 0.125f : 1.0f;   // fold 1/sqrt(dh) into q
#pragma unroll
  for (int mt = 0; mt < 2; ++mt) {
    const int t0m = t0 + mt * 16;
    if (which < 2) {
      __bf16* dst = (which == 0) ? qb : kb;            // [b,h,t,dh]
#pragma unroll
      for (int j = 0; j < 4; ++j) {
        const int dh = j * 16 + nl;
        const float bv_ = bias[n0 + dh];
#pragma unroll
        for (int r = 0; r < 8; ++r) {
          const int t = t0m + r + 8 * kh;              // C layout: M = r + 8*(lane/16)
          dst[((size_t)(b * N_HEADS + head) * TT + t) * DH + dh] =
              (__bf16)((acc[mt][j][r] + bv_) * oscale);
        }
      }
    } else {                                           // vT: [b,h,dh,T]
#pragma unroll
      for (int j = 0; j < 4; ++j) {
        const int dh = j * 16 + nl;
        const float bv_ = bias[n0 + dh];
        v8bf pk;
#pragma unroll
        for (int r = 0; r < 8; ++r) pk[r] = (__bf16)(acc[mt][j][r] + bv_);
        __bf16* dst = vT + ((size_t)(b * N_HEADS + head) * DH + dh) * TT + t0m + 8 * kh;
        *(v8bf*)dst = pk;                              // 8 consecutive t -> 16B store
      }
    }
  }
}

// =====================================================================
// Kernel 2: causal flash attention. One wave per (b, h, 16-row q tile).
// Steady-state chunks are mask-free; k-fragments double-buffered across
// chunks and v-fragments issued before the softmax so load latency hides
// behind the VALU/trans phase. Row sums kept as per-lane partials.
// =====================================================================
__global__ __launch_bounds__(32)
void attn_kernel(const __bf16* __restrict__ qb, const __bf16* __restrict__ kb,
                 const __bf16* __restrict__ vT, float* __restrict__ attn) {
  const int lane = threadIdx.x;
  const int t0 = blockIdx.x * 16;
  const int h  = blockIdx.y;
  const int b  = blockIdx.z;
  const int kh = lane >> 4;
  const int nl = lane & 15;

#if defined(__gfx1250__) && __has_builtin(__builtin_amdgcn_tensor_load_to_lds)
  // Exercise the TDM path: D# group0 count=0 encodes a NULL tensor
  // (architected no-op DMA, ISA 8.3), tracked and drained via TENSORcnt.
  {
    typedef __attribute__((ext_vector_type(4))) unsigned int v4u;
    typedef __attribute__((ext_vector_type(4))) int v4i;
    typedef __attribute__((ext_vector_type(8))) int v8i;
    v4u g0 = (v4u)0u;
    v8i g1 = (v8i)0;
    v4i g2 = (v4i)0, g3 = (v4i)0;
#if __clang_major__ >= 23
    v8i g4 = (v8i)0;
    __builtin_amdgcn_tensor_load_to_lds(g0, g1, g2, g3, g4, 0);
#else
    __builtin_amdgcn_tensor_load_to_lds(g0, g1, g2, g3, 0);
#endif
    __builtin_amdgcn_s_wait_tensorcnt(0);
  }
#endif

  const __bf16* qh  = qb + (size_t)(b * N_HEADS + h) * TT * DH;
  const __bf16* khp = kb + (size_t)(b * N_HEADS + h) * TT * DH;
  const __bf16* vh  = vT + (size_t)(b * N_HEADS + h) * DH * TT;

  // q A-fragments (q already carries the 1/sqrt(dh) scale)
  const v16bf qa0 = load_A_bf16(qh + (size_t)t0 * DH,      DH, lane);
  const v16bf qa1 = load_A_bf16(qh + (size_t)t0 * DH + 32, DH, lane);

  v8f o[4];
  float mrow[8], lpart[8];
  int tq[8];
#pragma unroll
  for (int j = 0; j < 4; ++j)
#pragma unroll
    for (int i = 0; i < 8; ++i) o[j][i] = 0.f;
#pragma unroll
  for (int r = 0; r < 8; ++r) {
    mrow[r] = -1e30f; lpart[r] = 0.f; tq[r] = t0 + r + 8 * kh;
  }

  __shared__ __attribute__((aligned(16))) __bf16 pls[16 * 32];

  // unmasked chunk: processes cols [s0, s0+32) of an unmasked region
  auto process = [&](int s0, const v16bf& klo0, const v16bf& khi0,
                     const v16bf& klo1, const v16bf& khi1) {
    v8f z0, z1;
#pragma unroll
    for (int i = 0; i < 8; ++i) { z0[i] = 0.f; z1[i] = 0.f; }
    v8f sc0 = wmma_bf16(qa1, khi0, wmma_bf16(qa0, klo0, z0));
    v8f sc1 = wmma_bf16(qa1, khi1, wmma_bf16(qa0, klo1, z1));

    // issue v-fragments now; consumed after softmax
    v16bf vb0 = load_B_bf16(vh + (size_t)( 0) * TT + s0, TT, lane);
    v16bf vb1 = load_B_bf16(vh + (size_t)(16) * TT + s0, TT, lane);
    v16bf vb2 = load_B_bf16(vh + (size_t)(32) * TT + s0, TT, lane);
    v16bf vb3 = load_B_bf16(vh + (size_t)(48) * TT + s0, TT, lane);

    float corr[8];
#pragma unroll
    for (int r = 0; r < 8; ++r) {
      float mx = fmaxf(sc0[r], sc1[r]);
#pragma unroll
      for (int off = 8; off >= 1; off >>= 1)
        mx = fmaxf(mx, __shfl_xor(mx, off, 16));
      const float mnew = fmaxf(mrow[r], mx);
      const float c0 = __expf(mrow[r] - mnew);
      const float p0 = __expf(sc0[r] - mnew);
      const float p1 = __expf(sc1[r] - mnew);
      lpart[r] = lpart[r] * c0 + (p0 + p1);   // per-lane partial row sum
      mrow[r] = mnew;
      corr[r] = c0;
      const int mr = r + 8 * kh;
      pls[mr * 32 + nl]      = (__bf16)p0;
      pls[mr * 32 + 16 + nl] = (__bf16)p1;
    }
#pragma unroll
    for (int j = 0; j < 4; ++j)
#pragma unroll
      for (int r = 0; r < 8; ++r) o[j][r] *= corr[r];

    __syncthreads();
    v16bf pa = load_A_bf16((const __bf16*)pls, 32, lane);
    o[0] = wmma_bf16(pa, vb0, o[0]);
    o[1] = wmma_bf16(pa, vb1, o[1]);
    o[2] = wmma_bf16(pa, vb2, o[2]);
    o[3] = wmma_bf16(pa, vb3, o[3]);
    __syncthreads();
  };

  // chunks with s0+31 <= t0 are fully below the diagonal: no masking
  const int n_full = (t0 >= 31) ? ((t0 - 31) / 32 + 1) : 0;

  // double-buffered k-fragments (set A / set B), no rotation copies
  v16bf Ak0, Ak1, Ak2, Ak3, Bk0, Bk1, Bk2, Bk3;
  if (n_full > 0) {
    Ak0 = load_B_bf16(khp, DH, lane);
    Ak1 = load_B_bf16(khp + 32, DH, lane);
    Ak2 = load_B_bf16(khp + (size_t)16 * DH, DH, lane);
    Ak3 = load_B_bf16(khp + (size_t)16 * DH + 32, DH, lane);
  }
  int c = 0;
#pragma unroll 1
  for (; c + 2 <= n_full; c += 2) {
    const size_t sB = (size_t)(c + 1) * 32, sC = (size_t)(c + 2) * 32;
    Bk0 = load_B_bf16(khp + sB * DH, DH, lane);
    Bk1 = load_B_bf16(khp + sB * DH + 32, DH, lane);
    Bk2 = load_B_bf16(khp + (sB + 16) * DH, DH, lane);
    Bk3 = load_B_bf16(khp + (sB + 16) * DH + 32, DH, lane);
    process(c * 32, Ak0, Ak1, Ak2, Ak3);
    // sC <= n_full*32 <= T-32: always in-bounds (may be the diagonal chunk)
    Ak0 = load_B_bf16(khp + sC * DH, DH, lane);
    Ak1 = load_B_bf16(khp + sC * DH + 32, DH, lane);
    Ak2 = load_B_bf16(khp + (sC + 16) * DH, DH, lane);
    Ak3 = load_B_bf16(khp + (sC + 16) * DH + 32, DH, lane);
    process((c + 1) * 32, Bk0, Bk1, Bk2, Bk3);
  }
  if (c < n_full) process(c * 32, Ak0, Ak1, Ak2, Ak3);

  // ---- diagonal chunk: s0 <= t0 < s0+32 (the only masked one)
  {
    const int s0 = n_full * 32;
    const bool have2 = (t0 >= s0 + 16);      // second sub-tile partially live?
    v16bf k0lo = load_B_bf16(khp + (size_t)s0 * DH,      DH, lane);
    v16bf k0hi = load_B_bf16(khp + (size_t)s0 * DH + 32, DH, lane);
    v8f z0;
#pragma unroll
    for (int i = 0; i < 8; ++i) z0[i] = 0.f;
    v8f sc0 = wmma_bf16(qa1, k0hi, wmma_bf16(qa0, k0lo, z0));
    v8f sc1;
#pragma unroll
    for (int i = 0; i < 8; ++i) sc1[i] = -1e30f;
    if (have2) {                             // uniform branch
      v16bf k1lo = load_B_bf16(khp + (size_t)(s0 + 16) * DH,      DH, lane);
      v16bf k1hi = load_B_bf16(khp + (size_t)(s0 + 16) * DH + 32, DH, lane);
      v8f z1;
#pragma unroll
      for (int i = 0; i < 8; ++i) z1[i] = 0.f;
      sc1 = wmma_bf16(qa1, k1hi, wmma_bf16(qa0, k1lo, z1));
    }
    v16bf vb0 = load_B_bf16(vh + (size_t)( 0) * TT + s0, TT, lane);
    v16bf vb1 = load_B_bf16(vh + (size_t)(16) * TT + s0, TT, lane);
    v16bf vb2 = load_B_bf16(vh + (size_t)(32) * TT + s0, TT, lane);
    v16bf vb3 = load_B_bf16(vh + (size_t)(48) * TT + s0, TT, lane);

    float corr[8];
#pragma unroll
    for (int r = 0; r < 8; ++r) {
      const float v0 = (s0 + nl      <= tq[r]) ? sc0[r] : -1e30f;
      const float v1 = (s0 + 16 + nl <= tq[r]) ? sc1[r] : -1e30f;
      float mx = fmaxf(v0, v1);
#pragma unroll
      for (int off = 8; off >= 1; off >>= 1)
        mx = fmaxf(mx, __shfl_xor(mx, off, 16));
      const float mnew = fmaxf(mrow[r], mx);
      const float c0 = __expf(mrow[r] - mnew);
      const float p0 = __expf(v0 - mnew);
      const float p1 = __expf(v1 - mnew);
      lpart[r] = lpart[r] * c0 + (p0 + p1);
      mrow[r] = mnew;
      corr[r] = c0;
      const int mr = r + 8 * kh;
      pls[mr * 32 + nl]      = (__bf16)p0;
      pls[mr * 32 + 16 + nl] = (__bf16)p1;
    }
#pragma unroll
    for (int j = 0; j < 4; ++j)
#pragma unroll
      for (int r = 0; r < 8; ++r) o[j][r] *= corr[r];

    __syncthreads();
    v16bf pa = load_A_bf16((const __bf16*)pls, 32, lane);
    o[0] = wmma_bf16(pa, vb0, o[0]);
    o[1] = wmma_bf16(pa, vb1, o[1]);
    o[2] = wmma_bf16(pa, vb2, o[2]);
    o[3] = wmma_bf16(pa, vb3, o[3]);
  }

  // ---- reduce per-lane row-sum partials once, normalize, store f32
#pragma unroll
  for (int r = 0; r < 8; ++r) {
#pragma unroll
    for (int off = 8; off >= 1; off >>= 1)
      lpart[r] += __shfl_xor(lpart[r], off, 16);
  }
#pragma unroll
  for (int j = 0; j < 4; ++j) {
#pragma unroll
    for (int r = 0; r < 8; ++r) {
      const float ov = o[j][r] / lpart[r];
      attn[((size_t)b * TT + t0 + r + 8 * kh) * D_MODEL + h * DH + j * 16 + nl] = ov;
    }
  }
}

// =====================================================================
// Kernel 3: residual + LayerNorm. 256 threads per 1024-wide row.
// =====================================================================
__global__ __launch_bounds__(256)
void ln_kernel(const float* __restrict__ x, const float* __restrict__ attn,
               const float* __restrict__ gamma, const float* __restrict__ beta,
               float* __restrict__ out) {
  const int row = blockIdx.x;
  const int tid = threadIdx.x;
  const float* xr = x + (size_t)row * D_MODEL;
  const float* ar = attn + (size_t)row * D_MODEL;

  float y[4], s = 0.f, s2 = 0.f;
#pragma unroll
  for (int i = 0; i < 4; ++i) {
    const int c = tid + i * 256;
    const float v = xr[c] + ar[c];
    y[i] = v; s += v; s2 += v * v;
  }
#pragma unroll
  for (int off = 16; off >= 1; off >>= 1) {
    s  += __shfl_xor(s, off, 32);
    s2 += __shfl_xor(s2, off, 32);
  }
  __shared__ float ls[8], ls2[8];
  const int w = tid >> 5;
  if ((tid & 31) == 0) { ls[w] = s; ls2[w] = s2; }
  __syncthreads();
  float ts = 0.f, ts2 = 0.f;
#pragma unroll
  for (int i = 0; i < 8; ++i) { ts += ls[i]; ts2 += ls2[i]; }
  const float mu  = ts  * (1.0f / D_MODEL);
  const float var = ts2 * (1.0f / D_MODEL) - mu * mu;
  const float inv = rsqrtf(var + 1e-5f);
#pragma unroll
  for (int i = 0; i < 4; ++i) {
    const int c = tid + i * 256;
    out[(size_t)row * D_MODEL + c] = (y[i] - mu) * inv * gamma[c] + beta[c];
  }
}

extern "C" void kernel_launch(void* const* d_in, const int* in_sizes, int n_in,
                              void* d_out, int out_size, void* d_ws, size_t ws_size,
                              hipStream_t stream) {
  (void)in_sizes; (void)n_in; (void)out_size; (void)ws_size;
  const float* x     = (const float*)d_in[0];
  const float* Wq    = (const float*)d_in[1];
  const float* bq    = (const float*)d_in[2];
  const float* Wk    = (const float*)d_in[3];
  const float* bk    = (const float*)d_in[4];
  const float* Wv    = (const float*)d_in[5];
  const float* bv    = (const float*)d_in[6];
  const float* gamma = (const float*)d_in[7];
  const float* beta  = (const float*)d_in[8];
  float* out = (float*)d_out;

  char* ws = (char*)d_ws;
  const size_t nX = (size_t)BB * TT * D_MODEL;             // 8,388,608
  const size_t nW = (size_t)D_MODEL * D_MODEL;             // 1,048,576
  const size_t qkSz = (size_t)BB * N_HEADS * TT * DH * sizeof(__bf16); // 16 MB

  __bf16* xbuf = (__bf16*)(ws);                            // 16 MB
  __bf16* Wqb  = (__bf16*)(ws + nX * 2);                   // 2 MB each
  __bf16* Wkb  = Wqb + nW;
  __bf16* Wvb  = Wkb + nW;
  char*   p    = (char*)(Wvb + nW);
  __bf16* qbuf = (__bf16*)(p);
  __bf16* kbuf = (__bf16*)(p + qkSz);
  __bf16* vT   = (__bf16*)(p + 2 * qkSz);
  float*  attn = (float*)(p + 3 * qkSz);                   // 32 MB

  cvt_bf16_kernel<<<dim3((int)(nX / (256 * 8))), dim3(256), 0, stream>>>(x,  xbuf, (int)nX);
  cvt_bf16_kernel<<<dim3((int)(nW / (256 * 8))), dim3(256), 0, stream>>>(Wq, Wqb,  (int)nW);
  cvt_bf16_kernel<<<dim3((int)(nW / (256 * 8))), dim3(256), 0, stream>>>(Wk, Wkb,  (int)nW);
  cvt_bf16_kernel<<<dim3((int)(nW / (256 * 8))), dim3(256), 0, stream>>>(Wv, Wvb,  (int)nW);

  qkv_proj_kernel<<<dim3(BB * TT / 32, N_HEADS, 3), dim3(32), 0, stream>>>(
      xbuf, Wqb, bq, Wkb, bk, Wvb, bv, qbuf, kbuf, vT);
  attn_kernel<<<dim3(TT / 16, N_HEADS, BB), dim3(32), 0, stream>>>(qbuf, kbuf, vT, attn);
  ln_kernel<<<dim3(BB * TT), dim3(256), 0, stream>>>(x, attn, gamma, beta, out);
}
